// ConvLSTM2D_Code_27109833572862
// MI455X (gfx1250) — matrix operand order repaired and checked by reading, not compiled, not run on hardware
//
#include <hip/hip_runtime.h>
#include <hip/hip_fp16.h>
#include <math.h>

typedef __attribute__((ext_vector_type(16))) _Float16 v16h;
typedef __attribute__((ext_vector_type(8)))  _Float16 v8h;
typedef __attribute__((ext_vector_type(8)))  float    v8f;

#define B_   8
#define T_   10
#define H_   64
#define W_   64
#define CIN_ 32
#define F_   64
#define NCH_ 256   // 4*F
#define G_   4

#define WX_PACK_ELEMS (9*16*32*16)            // 73728 halfs
#define WH_PACK_ELEMS (G_*9*2*16*32*16)       // 589824 halfs
#define HWF_  (H_*W_*F_)                      // 262144
#define BHWF_ (B_*HWF_)                       // 2097152

// workspace byte offsets (256B aligned)
#define WS_WX_OFF  0u
#define WS_WH_OFF  147456u                    // WX_PACK_ELEMS*2
#define WS_H_OFF   1327104u                   // + WH_PACK_ELEMS*2
#define WS_C_OFF   9715712u                   // + 2*BHWF_*2

// ---------------------------------------------------------------------------
// Pack f32 weights into f16 WMMA B-fragment order.
// B frag order used here (contiguous K, same as A): lane l holds column n = l%16,
// 16 contiguous halfs j=0..15 -> K = (l<16 ? 0 : 16) + j.
// ---------------------------------------------------------------------------
__global__ void pack_weights_kernel(const float* __restrict__ kern,
                                    const float* __restrict__ rkern,
                                    _Float16* __restrict__ wx,
                                    _Float16* __restrict__ wh) {
    int i = blockIdx.x * blockDim.x + threadIdx.x;
    if (i < WX_PACK_ELEMS) {
        int j     = i & 15;
        int lane  = (i >> 4) & 31;
        int ntile = (i >> 9) & 15;
        int tap   = i >> 13;                       // 0..8
        int k     = ((lane < 16) ? 0 : 16) + j;    // cin 0..31
        int n     = ntile * 16 + (lane & 15);
        wx[i] = (_Float16)kern[(tap * CIN_ + k) * NCH_ + n];
    }
    if (i < WH_PACK_ELEMS) {
        int j     = i & 15;
        int lane  = (i >> 4) & 31;
        int ntile = (i >> 9) & 15;
        int kc    = (i >> 13) & 1;
        int gt    = i >> 14;                       // g*9 + tap
        int tap   = gt % 9;
        int g     = gt / 9;
        int klocal = kc * 32 + ((lane < 16) ? 0 : 16) + j;   // cin 0..63
        int n     = ntile * 16 + (lane & 15);
        wh[i] = (_Float16)rkern[(size_t)((g * 9 + tap) * F_ + klocal) * NCH_ + n];
    }
}

__global__ void init_state_kernel(float* __restrict__ c, _Float16* __restrict__ h0) {
    int i = blockIdx.x * blockDim.x + threadIdx.x;
    if (i < BHWF_) { c[i] = 0.0f; h0[i] = (_Float16)0.0f; }
}

// ---------------------------------------------------------------------------
// One ConvLSTM timestep. Block = 256 threads (8 wave32s).
// Block tile: 32 pixels (half a W row) x 256 gate channels.
// wave_m = wid&1 -> 16-pixel M tile, wave_n = wid>>1 -> 64-channel N range.
// Unified 27-step K loop (9 input taps + 9x2 recurrent tap/kchunk), software
// pipelined: next step's A (LDS) and B (global/L2) fragments are prefetched
// into registers while the current step's 4 WMMAs execute.
// ---------------------------------------------------------------------------
__global__ __launch_bounds__(256) void lstm_step_kernel(
        const float*    __restrict__ x,       // (B,T,H,W,CIN)
        const int*      __restrict__ labels,  // (B,)
        const float*    __restrict__ bias,    // (G,256)
        const _Float16* __restrict__ wx,      // packed
        const _Float16* __restrict__ wh,      // packed
        const _Float16* __restrict__ h_in,    // (B,H,W,F) f16
        _Float16*       __restrict__ h_out,   // (B,H,W,F) f16
        float*          __restrict__ c_state, // (B,H,W,F) f32
        float*          __restrict__ out,     // (B,T,H,W,F) f32
        int t)
{
    __shared__ alignas(16) _Float16 xs[3][34][CIN_];   // 6.4 KB
    __shared__ alignas(16) _Float16 hs[3][34][F_];     // 12.8 KB
    __shared__ float zbuf[32][NCH_];                   // 32 KB

    const int blk   = blockIdx.x;
    const int wtile = blk & 1;
    const int row   = (blk >> 1) & 63;
    const int b     = blk >> 7;
    const int wbase = wtile * 32;

    const int tid  = threadIdx.x;
    const int lane = tid & 31;
    const int wid  = tid >> 5;
    const int lbl  = labels[b];

    // ---- stage x halo rows (f32 -> f16) ----
    for (int i = tid; i < 3 * 34 * CIN_; i += 256) {
        int c   = i & (CIN_ - 1);
        int col = (i >> 5) % 34;
        int dy  = i / (34 * CIN_);
        int gy = row + dy - 1;
        int gx = wbase + col - 1;
        float v = 0.0f;
        if (gy >= 0 && gy < H_ && gx >= 0 && gx < W_)
            v = x[(size_t)(((b * T_ + t) * H_ + gy) * W_ + gx) * CIN_ + c];
        xs[dy][col][c] = (_Float16)v;
    }
    // ---- stage h_prev halo rows (already f16) ----
    for (int i = tid; i < 3 * 34 * F_; i += 256) {
        int c   = i & (F_ - 1);
        int col = (i >> 6) % 34;
        int dy  = i / (34 * F_);
        int gy = row + dy - 1;
        int gx = wbase + col - 1;
        _Float16 v = (_Float16)0.0f;
        if (gy >= 0 && gy < H_ && gx >= 0 && gx < W_)
            v = h_in[(size_t)((b * H_ + gy) * W_ + gx) * F_ + c];
        hs[dy][col][c] = v;
    }
    __syncthreads();

    const int wave_m  = wid & 1;
    const int wave_n  = wid >> 1;                // 0..3 -> 64-ch range
    const int local_w = wave_m * 16 + (lane & 15);
    const int c0      = (lane < 16) ? 0 : 16;    // A frag K sub-base

    const _Float16* whl = wh + (size_t)lbl * (9 * 2 * 16 * 32 * 16);

    union AB { v16h v; v8h h[2]; };

    // A fragment for unified K-step s (0..8 input, 9..26 recurrent)
    auto load_a = [&](int s) -> v16h {
        AB a;
        if (s < 9) {
            int dy = s / 3, dx = s % 3;
            a.h[0] = *(const v8h*)&xs[dy][local_w + dx][c0];
            a.h[1] = *(const v8h*)&xs[dy][local_w + dx][c0 + 8];
        } else {
            int r = s - 9, tap = r >> 1, kc = r & 1;
            int dy = tap / 3, dx = tap % 3;
            a.h[0] = *(const v8h*)&hs[dy][local_w + dx][kc * 32 + c0];
            a.h[1] = *(const v8h*)&hs[dy][local_w + dx][kc * 32 + c0 + 8];
        }
        return a.v;
    };
    // B fragment for step s, local N-tile nt
    auto load_b = [&](int s, int nt) -> v16h {
        const _Float16* p;
        if (s < 9) {
            p = wx + (size_t)(((s * 16 + wave_n * 4 + nt) * 32 + lane) * 16);
        } else {
            int r = s - 9;   // == tap*2 + kc, matches packed layout
            p = whl + (size_t)(((r * 16 + wave_n * 4 + nt) * 32 + lane) * 16);
        }
        return *(const v16h*)p;
    };

    v8f acc[4] = {};

    // ---- software-pipelined K loop: 27 steps x 4 WMMAs ----
    v16h a_cur = load_a(0);
    v16h b_cur[4];
#pragma unroll
    for (int nt = 0; nt < 4; ++nt) b_cur[nt] = load_b(0, nt);

#pragma unroll
    for (int s = 0; s < 27; ++s) {
        v16h a_nxt = {};
        v16h b_nxt[4] = {};
        if (s < 26) {
            a_nxt = load_a(s + 1);
#pragma unroll
            for (int nt = 0; nt < 4; ++nt) b_nxt[nt] = load_b(s + 1, nt);
        }
#pragma unroll
        for (int nt = 0; nt < 4; ++nt)
            acc[nt] = __builtin_amdgcn_wmma_f32_16x16x32_f16(
                false, a_cur, false, b_cur[nt], (short)0, acc[nt], false, false);
        a_cur = a_nxt;
#pragma unroll
        for (int nt = 0; nt < 4; ++nt) b_cur[nt] = b_nxt[nt];
    }

    // ---- park z in LDS (C-frag layout: VGPR r, lane<16 -> M=r else M=r+8) ----
#pragma unroll
    for (int nt = 0; nt < 4; ++nt) {
        int n = wave_n * 64 + nt * 16 + (lane & 15);
#pragma unroll
        for (int r = 0; r < 8; ++r) {
            int m = wave_m * 16 + ((lane < 16) ? r : r + 8);
            zbuf[m][n] = acc[nt][r];
        }
    }
    __syncthreads();

    // ---- LSTM gates: 32 pixels x 64 feature channels ----
    const float* bptr = bias + lbl * NCH_;
    for (int i = tid; i < 32 * F_; i += 256) {
        int m  = i >> 6;
        int fc = i & (F_ - 1);
        float zi = zbuf[m][fc]        + bptr[fc];
        float zf = zbuf[m][64 + fc]   + bptr[64 + fc];
        float zg = zbuf[m][128 + fc]  + bptr[128 + fc];
        float zo = zbuf[m][192 + fc]  + bptr[192 + fc];
        float ig = 1.0f / (1.0f + __expf(-zi));
        float fg = 1.0f / (1.0f + __expf(-zf));
        float gg = tanhf(zg);
        float og = 1.0f / (1.0f + __expf(-zo));
        int gw = wbase + m;
        size_t sidx = (size_t)((b * H_ + row) * W_ + gw) * F_ + fc;
        float cn = fg * c_state[sidx] + ig * gg;
        c_state[sidx] = cn;
        float hn = og * tanhf(cn);
        h_out[sidx] = (_Float16)hn;
        // write-once 84MB output stream: non-temporal store, keep L2 for
        // weights + halo reuse
        __builtin_nontemporal_store(
            hn, &out[(size_t)(((b * T_ + t) * H_ + row) * W_ + gw) * F_ + fc]);
    }
}

// ---------------------------------------------------------------------------
extern "C" void kernel_launch(void* const* d_in, const int* in_sizes, int n_in,
                              void* d_out, int out_size, void* d_ws, size_t ws_size,
                              hipStream_t stream) {
    (void)in_sizes; (void)n_in; (void)out_size; (void)ws_size;

    const float* x      = (const float*)d_in[0];   // inputs
    const int*   labels = (const int*)  d_in[1];   // label_values
    const float* kern   = (const float*)d_in[2];   // kernel
    const float* rkern  = (const float*)d_in[3];   // recurrent_kernel
    const float* bias   = (const float*)d_in[4];   // bias
    float*       out    = (float*)d_out;

    char* ws = (char*)d_ws;
    _Float16* wx   = (_Float16*)(ws + WS_WX_OFF);
    _Float16* wh   = (_Float16*)(ws + WS_WH_OFF);
    _Float16* hbuf = (_Float16*)(ws + WS_H_OFF);     // 2 * BHWF_ halfs (ping-pong)
    float*    cbuf = (float*)   (ws + WS_C_OFF);     // BHWF_ floats

    pack_weights_kernel<<<(WH_PACK_ELEMS + 255) / 256, 256, 0, stream>>>(kern, rkern, wx, wh);
    init_state_kernel<<<(BHWF_ + 255) / 256, 256, 0, stream>>>(cbuf, hbuf);

    for (int t = 0; t < T_; ++t) {
        const _Float16* h_in  = hbuf + (size_t)(t & 1) * BHWF_;
        _Float16*       h_out = hbuf + (size_t)((t + 1) & 1) * BHWF_;
        lstm_step_kernel<<<B_ * H_ * 2, 256, 0, stream>>>(
            x, labels, bias, wx, wh, h_in, h_out, cbuf, out, t);
    }
}
